// DeepseekV3Attention_71657234367034
// MI455X (gfx1250) — matrix operand order repaired and checked by reading, not compiled, not run on hardware
//
#include <hip/hip_runtime.h>
#include <cstdint>
#include <cstddef>

// ---------------- problem constants (DeepSeek V3 attn, B=1) ----------------
#define S_LEN   2048
#define H_DIM   2048
#define QL_DIM  1536
#define KVL_DIM 512
#define NH_N    16
#define NOPE_D  128
#define ROPE_D  64
#define VD_D    128
#define IH_N    16
#define ID_D    128
#define QKD     192      // NOPE + ROPE
#define QB_DIM  3072     // NH * (NOPE+ROPE)
#define KVB_DIM 4096     // NH * (NOPE+VD)
#define TOPK_N  512
#define EPSF    1e-6f

// ---------------- types ----------------
typedef __bf16 bf16_t;
typedef bf16_t v16bf __attribute__((ext_vector_type(16)));
typedef bf16_t v8bf  __attribute__((ext_vector_type(8)));
typedef float  v8f   __attribute__((ext_vector_type(8)));

__device__ __forceinline__ unsigned short f2bf(float f) {
  union { float f; unsigned u; } v; v.f = f;
  unsigned r = v.u + 0x7FFFu + ((v.u >> 16) & 1u);   // RNE
  return (unsigned short)(r >> 16);
}

__device__ __forceinline__ v16bf pack16(v8bf lo, v8bf hi) {
  v16bf r;
#pragma unroll
  for (int i = 0; i < 8; ++i) { r[i] = lo[i]; r[8 + i] = hi[i]; }
  return r;
}

// ---------------- async global->LDS staging (CDNA5) ----------------
// GLOBAL_LOAD_ASYNC_TO_LDS_B128: ASYNCcnt-tracked DMA into LDS, no VGPR data.
// Builtin signature (from compiler diagnostic): (v4i AS1*, v4i AS3*, Imm, Imm).
#if defined(__HIP_DEVICE_COMPILE__) &&                                   \
    __has_builtin(__builtin_amdgcn_global_load_async_to_lds_b128) &&     \
    __has_builtin(__builtin_amdgcn_s_wait_asynccnt)
#define ASYNC_LDS 1
typedef int v4i_t __attribute__((ext_vector_type(4)));
typedef __attribute__((address_space(1))) v4i_t* gv4p;   // global b128 ptr
typedef __attribute__((address_space(3))) v4i_t* lv4p;   // LDS b128 ptr
#else
#define ASYNC_LDS 0
#endif

// copy 16 bf16 (32B) from global to LDS
__device__ __forceinline__ void copy16(unsigned short* lds_dst,
                                       const unsigned short* gsrc) {
#if ASYNC_LDS
  __builtin_amdgcn_global_load_async_to_lds_b128((gv4p)(gsrc),
                                                 (lv4p)(lds_dst), 0, 0);
  __builtin_amdgcn_global_load_async_to_lds_b128((gv4p)(gsrc + 8),
                                                 (lv4p)(lds_dst + 8), 0, 0);
#else
  const uint4* g = (const uint4*)gsrc;
  uint4 a = g[0], b = g[1];
  uint4* l = (uint4*)lds_dst;
  l[0] = a; l[1] = b;
#endif
}

__device__ __forceinline__ void wait_async_tiles() {
#if ASYNC_LDS
  __builtin_amdgcn_s_wait_asynccnt(0);
#endif
}

// =====================================================================
// Generic NT GEMM:  C[M][N] = alpha * A[M][K] * B[N][K]^T
// A, B: bf16 (as u16), C: f32.  M multiple of 128, K multiple of 32.
// N arbitrary (B rows clamped; OOB columns never stored).
// Tile 128x128, 8 waves (wave32), each 32x64.
// Double-buffered LDS; async global->LDS staging when available.
// Core: v_wmma_f32_16x16x32_bf16.
// =====================================================================
#define GTK  32
#define ALDS 40   // elems per A row (80B, 16B aligned)
#define BLDS 48   // elems per B row (96B, 32B aligned)

__global__ __launch_bounds__(256)
void gemm_nt_bf16(const unsigned short* __restrict__ A,
                  const unsigned short* __restrict__ B,
                  float* __restrict__ C,
                  int M, int N, int K,
                  int lda, int ldb, int ldc, float alpha) {
  __shared__ __attribute__((aligned(32))) unsigned short As[2][128 * ALDS];
  __shared__ __attribute__((aligned(32))) unsigned short Bs[2][128 * BLDS];
  (void)M;

  const int tid  = threadIdx.x;
  const int lane = tid & 31;
  const int wid  = tid >> 5;
  const int m0 = blockIdx.y * 128;
  const int n0 = blockIdx.x * 128;
  const int wm = (wid >> 1) * 32;
  const int wn = (wid & 1) * 64;

  v8f zero = {};
  v8f acc[2][4];
#pragma unroll
  for (int i = 0; i < 2; ++i)
#pragma unroll
    for (int j = 0; j < 4; ++j) acc[i][j] = zero;

  const int lrow = tid >> 1;
  const int lcol = (tid & 1) * 16;
  const int brow = (n0 + lrow < N) ? (n0 + lrow) : (N - 1);  // clamp
  const unsigned short* Ab = A + (size_t)(m0 + lrow) * lda + lcol;
  const unsigned short* Bb = B + (size_t)brow * ldb + lcol;
  unsigned short* Asl0 = &As[0][lrow * ALDS + lcol];
  unsigned short* Asl1 = &As[1][lrow * ALDS + lcol];
  unsigned short* Bsl0 = &Bs[0][lrow * BLDS + lcol];
  unsigned short* Bsl1 = &Bs[1][lrow * BLDS + lcol];

  // preload tile 0
  copy16(Asl0, Ab);
  copy16(Bsl0, Bb);

  const int fr = lane & 15;   // M (or N) index within 16x16 tile
  const int kh = lane >> 4;   // which K half this lane holds

  int buf = 0;
  for (int k0 = 0; k0 < K; k0 += GTK) {
    wait_async_tiles();
    __syncthreads();   // tile[buf] ready for all waves; prev compute finished
    if (k0 + GTK < K) {
      copy16(buf ? Asl0 : Asl1, Ab + k0 + GTK);
      copy16(buf ? Bsl0 : Bsl1, Bb + k0 + GTK);
    }

    const unsigned short* Asb = &As[buf][0];
    const unsigned short* Bsb = &Bs[buf][0];
    v16bf af[2];
#pragma unroll
    for (int mt = 0; mt < 2; ++mt) {
      const unsigned short* p = Asb + (wm + mt * 16 + fr) * ALDS + kh * 8;
      af[mt] = pack16(*(const v8bf*)p, *(const v8bf*)(p + 16));
    }
    v16bf bfm[4];
#pragma unroll
    for (int nt = 0; nt < 4; ++nt)
      bfm[nt] = pack16(*(const v8bf*)(Bsb + (wn + nt * 16 + fr) * BLDS + kh * 8), *(const v8bf*)(Bsb + (wn + nt * 16 + fr) * BLDS + kh * 8 + 16));
#pragma unroll
    for (int mt = 0; mt < 2; ++mt)
#pragma unroll
      for (int nt = 0; nt < 4; ++nt)
        acc[mt][nt] = __builtin_amdgcn_wmma_f32_16x16x32_bf16(
            false, af[mt], false, bfm[nt], (short)0, acc[mt][nt], false, false);
    buf ^= 1;
  }

  // C/D layout: VGPR r -> M = r + 8*(lane>=16), N = lane&15
  const int cn = lane & 15;
  const int ch = lane >> 4;
#pragma unroll
  for (int mt = 0; mt < 2; ++mt)
#pragma unroll
    for (int nt = 0; nt < 4; ++nt) {
      int n = n0 + wn + nt * 16 + cn;
      if (n < N) {
#pragma unroll
        for (int r = 0; r < 8; ++r) {
          int m = m0 + wm + mt * 16 + r + 8 * ch;
          C[(size_t)m * ldc + n] = alpha * acc[mt][nt][r];
        }
      }
    }
}

// =====================================================================
// Fused indexer scores:
//   scores[t][s] = sum_h w[t][h] * relu(Qi[h][t][:] . Ki[s][:]) * ID^-0.5
//   causal (-inf above diagonal). Tile 128(t) x 64(s), WMMA per head.
// =====================================================================
__global__ __launch_bounds__(256)
void idx_scores_kernel(const unsigned short* __restrict__ Qi,  // [IH][S][ID]
                       const unsigned short* __restrict__ Ki,  // [S][ID]
                       const float* __restrict__ w,            // [S][IH]
                       float* __restrict__ scores)             // [S][S]
{
  __shared__ __attribute__((aligned(32))) unsigned short Qs[128 * ID_D];
  __shared__ __attribute__((aligned(32))) unsigned short Ks[64 * ID_D];
  __shared__ float wl[128];

  const int tid = threadIdx.x, lane = tid & 31, wid = tid >> 5;
  const int t0 = blockIdx.y * 128, s0 = blockIdx.x * 64;
  const int wm = (wid >> 1) * 32, wn = (wid & 1) * 32;

  { // K tile: 64 rows x 128 cols bf16 (async staged)
    int row = tid >> 2, col = (tid & 3) * 32;
    const unsigned short* kg = Ki + (size_t)(s0 + row) * ID_D + col;
    unsigned short* ks = Ks + row * ID_D + col;
    copy16(ks, kg);
    copy16(ks + 16, kg + 16);
  }

  v8f zero = {};
  v8f accS[2][2];
#pragma unroll
  for (int i = 0; i < 2; ++i)
#pragma unroll
    for (int j = 0; j < 2; ++j) accS[i][j] = zero;

  const float inv_sqrt_id = 0.08838834764831845f;  // 128^-0.5
  const int fr = lane & 15, kh = lane >> 4, ch = lane >> 4;
  const int qrow = tid >> 1, qcol = (tid & 1) * 64;

  for (int h = 0; h < IH_N; ++h) {
    __syncthreads();   // previous head's compute done before Qs overwrite
    { // Q tile for this head: 128 x 128 (async staged)
      const unsigned short* qg =
          Qi + ((size_t)h * S_LEN + t0 + qrow) * ID_D + qcol;
      unsigned short* qs = Qs + qrow * ID_D + qcol;
#pragma unroll
      for (int i = 0; i < 4; ++i) copy16(qs + i * 16, qg + i * 16);
    }
    if (tid < 128) wl[tid] = w[(size_t)(t0 + tid) * IH_N + h];
    wait_async_tiles();
    __syncthreads();

    v8f acc[2][2];
#pragma unroll
    for (int i = 0; i < 2; ++i)
#pragma unroll
      for (int j = 0; j < 2; ++j) acc[i][j] = zero;

#pragma unroll
    for (int k0 = 0; k0 < ID_D; k0 += 32) {
      v16bf af[2], bfm[2];
#pragma unroll
      for (int mt = 0; mt < 2; ++mt) {
        const unsigned short* p = Qs + (wm + mt * 16 + fr) * ID_D + k0 + kh * 8;
        af[mt] = pack16(*(const v8bf*)p, *(const v8bf*)(p + 16));
      }
#pragma unroll
      for (int nt = 0; nt < 2; ++nt)
        bfm[nt] = pack16(*(const v8bf*)(Ks + (wn + nt * 16 + fr) * ID_D + k0 + kh * 8), *(const v8bf*)(Ks + (wn + nt * 16 + fr) * ID_D + k0 + kh * 8 + 16));
#pragma unroll
      for (int mt = 0; mt < 2; ++mt)
#pragma unroll
        for (int nt = 0; nt < 2; ++nt)
          acc[mt][nt] = __builtin_amdgcn_wmma_f32_16x16x32_bf16(
              false, af[mt], false, bfm[nt], (short)0, acc[mt][nt], false, false);
    }
    // relu + per-row weight, accumulate over heads
#pragma unroll
    for (int mt = 0; mt < 2; ++mt)
#pragma unroll
      for (int nt = 0; nt < 2; ++nt)
#pragma unroll
        for (int r = 0; r < 8; ++r) {
          float v = acc[mt][nt][r];
          v = v > 0.f ? v : 0.f;
          accS[mt][nt][r] += v * wl[wm + mt * 16 + r + 8 * ch] * inv_sqrt_id;
        }
  }

  const int cn = lane & 15;
#pragma unroll
  for (int mt = 0; mt < 2; ++mt)
#pragma unroll
    for (int nt = 0; nt < 2; ++nt)
#pragma unroll
      for (int r = 0; r < 8; ++r) {
        int t = t0 + wm + mt * 16 + r + 8 * ch;
        int s = s0 + wn + nt * 16 + cn;
        scores[(size_t)t * S_LEN + s] = (s <= t) ? accS[mt][nt][r] : -3.0e38f;
      }
}

// =====================================================================
// Elementwise / normalization / build kernels
// =====================================================================
__global__ void cast_bf16_kernel(const float* __restrict__ in,
                                 unsigned short* __restrict__ out, int n) {
  int i = blockIdx.x * 256 + threadIdx.x;
  if (i < n) out[i] = f2bf(in[i]);
}

// in: f32 [K][N] row-major  ->  out: bf16 [N][K]
__global__ void transpose_cast_kernel(const float* __restrict__ in,
                                      unsigned short* __restrict__ out,
                                      int K, int N) {
  long i = (long)blockIdx.x * 256 + threadIdx.x;
  if (i < (long)K * N) {
    int n = (int)(i % N);
    int k = (int)(i / N);
    out[(size_t)n * K + k] = f2bf(in[i]);
  }
}

__global__ __launch_bounds__(256)
void rmsnorm_bf16_kernel(const float* __restrict__ in,
                         const float* __restrict__ gamma,
                         unsigned short* __restrict__ out,
                         int len, int istride, int ostride) {
  const int row = blockIdx.x, tid = threadIdx.x;
  const float* x = in + (size_t)row * istride;
  __shared__ float red[256];
  float ss = 0.f;
  for (int i = tid; i < len; i += 256) { float v = x[i]; ss += v * v; }
  red[tid] = ss; __syncthreads();
  for (int o = 128; o > 0; o >>= 1) {
    if (tid < o) red[tid] += red[tid + o];
    __syncthreads();
  }
  float r = rsqrtf(red[0] / (float)len + EPSF);
  for (int i = tid; i < len; i += 256)
    out[(size_t)row * ostride + i] = f2bf(x[i] * r * gamma[i]);
}

// Qcat[h][t][0:128]=q_nope ; [128:192]=rope(q_pe)
__global__ void build_q_kernel(const float* __restrict__ q,     // [S][3072]
                               const float* __restrict__ cosb,  // [S][32]
                               const float* __restrict__ sinb,
                               unsigned short* __restrict__ Qcat) {
  long i = (long)blockIdx.x * 256 + threadIdx.x;
  if (i >= (long)NH_N * S_LEN * QKD) return;
  int h = (int)(i / ((long)S_LEN * QKD));
  int rem = (int)(i % ((long)S_LEN * QKD));
  int t = rem / QKD, c = rem % QKD;
  const float* qp = q + (size_t)t * QB_DIM + h * QKD;
  float val;
  if (c < NOPE_D) val = qp[c];
  else {
    int j = c - NOPE_D;
    int ii = (j < 32) ? j : j - 32;
    float x1 = qp[NOPE_D + ii], x2 = qp[NOPE_D + 32 + ii];
    float cc = cosb[t * 32 + ii], sv = sinb[t * 32 + ii];
    val = (j < 32) ? (x1 * cc - x2 * sv) : (x1 * sv + x2 * cc);
  }
  Qcat[i] = f2bf(val);
}

// Kcat[h][s][0:128]=k_nope(kvb) ; [128:192]=rope(k_pe from kv)
__global__ void build_k_kernel(const float* __restrict__ kvb,  // [S][4096]
                               const float* __restrict__ kv,   // [S][576]
                               const float* __restrict__ cosb,
                               const float* __restrict__ sinb,
                               unsigned short* __restrict__ Kcat) {
  long i = (long)blockIdx.x * 256 + threadIdx.x;
  if (i >= (long)NH_N * S_LEN * QKD) return;
  int h = (int)(i / ((long)S_LEN * QKD));
  int rem = (int)(i % ((long)S_LEN * QKD));
  int s = rem / QKD, c = rem % QKD;
  float val;
  if (c < NOPE_D) val = kvb[(size_t)s * KVB_DIM + h * 256 + c];
  else {
    int j = c - NOPE_D;
    int ii = (j < 32) ? j : j - 32;
    const float* pe = kv + (size_t)s * 576 + KVL_DIM;
    float x1 = pe[ii], x2 = pe[32 + ii];
    float cc = cosb[s * 32 + ii], sv = sinb[s * 32 + ii];
    val = (j < 32) ? (x1 * cc - x2 * sv) : (x1 * sv + x2 * cc);
  }
  Kcat[i] = f2bf(val);
}

// Vt[h][d][s] = kvb[s][h*256 + 128 + d]   (transposed V for NT GEMM)
__global__ void build_vt_kernel(const float* __restrict__ kvb,
                                unsigned short* __restrict__ Vt) {
  long i = (long)blockIdx.x * 256 + threadIdx.x;
  if (i >= (long)NH_N * VD_D * S_LEN) return;
  int h = (int)(i / ((long)VD_D * S_LEN));
  int rem = (int)(i % ((long)VD_D * S_LEN));
  int d = rem / S_LEN, s = rem % S_LEN;
  Vt[i] = f2bf(kvb[(size_t)s * KVB_DIM + h * 256 + NOPE_D + d]);
}

// Qi[h][t][0:64]=rope(qi[:64]) ; [64:128]=qi[64:128]
__global__ void build_qi_kernel(const float* __restrict__ qi,  // [S][2048]
                                const float* __restrict__ cosb,
                                const float* __restrict__ sinb,
                                unsigned short* __restrict__ Qib) {
  long i = (long)blockIdx.x * 256 + threadIdx.x;
  if (i >= (long)IH_N * S_LEN * ID_D) return;
  int h = (int)(i / ((long)S_LEN * ID_D));
  int rem = (int)(i % ((long)S_LEN * ID_D));
  int t = rem / ID_D, c = rem % ID_D;
  const float* qp = qi + (size_t)t * (IH_N * ID_D) + h * ID_D;
  float val;
  if (c < ROPE_D) {
    int ii = (c < 32) ? c : c - 32;
    float x1 = qp[ii], x2 = qp[32 + ii];
    float cc = cosb[t * 32 + ii], sv = sinb[t * 32 + ii];
    val = (c < 32) ? (x1 * cc - x2 * sv) : (x1 * sv + x2 * cc);
  } else val = qp[c];
  Qib[i] = f2bf(val);
}

// LayerNorm + rope(first 64) for the indexer keys. One block (128 thr) per row.
__global__ __launch_bounds__(128)
void ln_rope_ki_kernel(const float* __restrict__ ki_f,  // [S][128]
                       const float* __restrict__ gamma,
                       const float* __restrict__ beta,
                       const float* __restrict__ cosb,
                       const float* __restrict__ sinb,
                       unsigned short* __restrict__ Ki) {
  const int s = blockIdx.x, i = threadIdx.x;
  __shared__ float xv[128];
  __shared__ float red[128];
  float v = ki_f[(size_t)s * ID_D + i];
  red[i] = v; __syncthreads();
  for (int o = 64; o > 0; o >>= 1) { if (i < o) red[i] += red[i + o]; __syncthreads(); }
  float mean = red[0] / 128.f; __syncthreads();
  float d = v - mean;
  red[i] = d * d; __syncthreads();
  for (int o = 64; o > 0; o >>= 1) { if (i < o) red[i] += red[i + o]; __syncthreads(); }
  float var = red[0] / 128.f;
  float ln = d * rsqrtf(var + EPSF) * gamma[i] + beta[i];
  xv[i] = ln; __syncthreads();
  float out;
  if (i < ROPE_D) {
    int ii = (i < 32) ? i : i - 32;
    float x1 = xv[ii], x2 = xv[32 + ii];
    float cc = cosb[s * 32 + ii], sv = sinb[s * 32 + ii];
    out = (i < 32) ? (x1 * cc - x2 * sv) : (x1 * sv + x2 * cc);
  } else out = xv[i];
  Ki[(size_t)s * ID_D + i] = f2bf(out);
}

// Top-k selection mask per row via 32-step bisection on sortable-uint keys.
__global__ __launch_bounds__(256)
void topk_mask_kernel(const float* __restrict__ scores,
                      unsigned char* __restrict__ mask, int topk) {
  const int t = blockIdx.x, tid = threadIdx.x;
  __shared__ int red[256];
  unsigned key[8]; int valid[8];
#pragma unroll
  for (int j = 0; j < 8; ++j) {
    int s = tid + j * 256;
    union { float f; unsigned u; } cv; cv.f = scores[(size_t)t * S_LEN + s];
    key[j] = (cv.u & 0x80000000u) ? ~cv.u : (cv.u | 0x80000000u);
    valid[j] = (s <= t) ? 1 : 0;
  }
  unsigned lo = 0u, hi = 0xFFFFFFFFu, thr = 0u;
  for (int it = 0; it < 32; ++it) {
    unsigned mid = lo + ((hi - lo) >> 1);
    int c = 0;
#pragma unroll
    for (int j = 0; j < 8; ++j) c += (valid[j] && key[j] >= mid) ? 1 : 0;
    red[tid] = c; __syncthreads();
    for (int o = 128; o > 0; o >>= 1) { if (tid < o) red[tid] += red[tid + o]; __syncthreads(); }
    int cnt = red[0]; __syncthreads();
    if (cnt >= topk) { thr = mid; if (mid == 0xFFFFFFFFu) break; lo = mid + 1; }
    else { if (mid == 0u) break; hi = mid - 1; }
    if (lo > hi) break;
  }
#pragma unroll
  for (int j = 0; j < 8; ++j) {
    int s = tid + j * 256;
    mask[(size_t)t * S_LEN + s] = (valid[j] && key[j] >= thr) ? 1 : 0;
  }
}

// Masked row softmax -> bf16 probabilities.
__global__ __launch_bounds__(256)
void softmax_mask_kernel(const float* __restrict__ sc,
                         const unsigned char* __restrict__ mask,
                         unsigned short* __restrict__ P) {
  const int t = blockIdx.x, tid = threadIdx.x;
  __shared__ float red[256];
  float v[8]; int m[8];
#pragma unroll
  for (int j = 0; j < 8; ++j) {
    int s = tid + j * 256;
    v[j] = sc[(size_t)t * S_LEN + s];
    m[j] = mask[(size_t)t * S_LEN + s];
  }
  float mx = -3.0e38f;
#pragma unroll
  for (int j = 0; j < 8; ++j) if (m[j] && v[j] > mx) mx = v[j];
  red[tid] = mx; __syncthreads();
  for (int o = 128; o > 0; o >>= 1) {
    if (tid < o) red[tid] = red[tid] > red[tid + o] ? red[tid] : red[tid + o];
    __syncthreads();
  }
  float gmax = red[0]; __syncthreads();
  float e[8]; float se = 0.f;
#pragma unroll
  for (int j = 0; j < 8; ++j) { e[j] = m[j] ? __expf(v[j] - gmax) : 0.f; se += e[j]; }
  red[tid] = se; __syncthreads();
  for (int o = 128; o > 0; o >>= 1) { if (tid < o) red[tid] += red[tid + o]; __syncthreads(); }
  float gs = red[0];
  float inv = gs > 0.f ? 1.f / gs : 0.f;
#pragma unroll
  for (int j = 0; j < 8; ++j)
    P[(size_t)t * S_LEN + tid + j * 256] = f2bf(e[j] * inv);
}

// =====================================================================
// Host launcher
// =====================================================================
extern "C" void kernel_launch(void* const* d_in, const int* in_sizes, int n_in,
                              void* d_out, int out_size, void* d_ws, size_t ws_size,
                              hipStream_t stream) {
  (void)in_sizes; (void)n_in; (void)out_size; (void)ws_size;
  const float* hidden   = (const float*)d_in[0];
  const float* cosb     = (const float*)d_in[1];
  const float* sinb     = (const float*)d_in[2];
  const float* Wq_a     = (const float*)d_in[3];
  const float* q_a_g    = (const float*)d_in[4];
  const float* Wq_b     = (const float*)d_in[5];
  const float* Wkv_a    = (const float*)d_in[6];
  const float* kv_a_g   = (const float*)d_in[7];
  const float* Wkv_b    = (const float*)d_in[8];
  const float* Wo       = (const float*)d_in[9];
  const float* Wq_idx   = (const float*)d_in[10];
  const float* Wk_idx   = (const float*)d_in[11];
  const float* Ww_idx   = (const float*)d_in[12];
  const float* kn_g     = (const float*)d_in[13];
  const float* kn_b     = (const float*)d_in[14];
  float* out_final = (float*)d_out;

  char* base = (char*)d_ws;
  size_t off = 0;
  auto alloc = [&](size_t bytes) -> void* {
    void* p = base + off;
    off = (off + bytes + 255) & ~((size_t)255);
    return p;
  };
  const size_t S = S_LEN;
  unsigned short* hidden_bf = (unsigned short*)alloc(S * H_DIM * 2);
  unsigned short* WqaT  = (unsigned short*)alloc((size_t)QL_DIM * H_DIM * 2);
  unsigned short* WqbT  = (unsigned short*)alloc((size_t)QB_DIM * QL_DIM * 2);
  unsigned short* WkvaT = (unsigned short*)alloc((size_t)576 * H_DIM * 2);
  unsigned short* WkvbT = (unsigned short*)alloc((size_t)KVB_DIM * KVL_DIM * 2);
  unsigned short* WoT   = (unsigned short*)alloc((size_t)H_DIM * (NH_N * VD_D) * 2);
  unsigned short* WqiT  = (unsigned short*)alloc((size_t)(IH_N * ID_D) * QL_DIM * 2);
  unsigned short* WkiT  = (unsigned short*)alloc((size_t)ID_D * H_DIM * 2);
  unsigned short* WwiT  = (unsigned short*)alloc((size_t)IH_N * H_DIM * 2);
  float* qa_f   = (float*)alloc(S * QL_DIM * 4);
  unsigned short* qa_bf = (unsigned short*)alloc(S * QL_DIM * 2);
  float* q_f    = (float*)alloc(S * QB_DIM * 4);
  float* kv_f   = (float*)alloc(S * 576 * 4);
  unsigned short* kvc_bf = (unsigned short*)alloc(S * KVL_DIM * 2);
  float* kvb_f  = (float*)alloc(S * KVB_DIM * 4);
  float* qi_f   = (float*)alloc(S * (IH_N * ID_D) * 4);
  float* ki_f   = (float*)alloc(S * ID_D * 4);
  float* w_f    = (float*)alloc(S * IH_N * 4);
  unsigned short* Qcat = (unsigned short*)alloc((size_t)NH_N * S * QKD * 2);
  unsigned short* Kcat = (unsigned short*)alloc((size_t)NH_N * S * QKD * 2);
  unsigned short* Vt   = (unsigned short*)alloc((size_t)NH_N * VD_D * S * 2);
  unsigned short* Qib  = (unsigned short*)alloc((size_t)IH_N * S * ID_D * 2);
  unsigned short* Kib  = (unsigned short*)alloc(S * ID_D * 2);
  float* idx_sc = (float*)alloc(S * S * 4);
  unsigned char* mask = (unsigned char*)alloc(S * S);
  float* sc_tmp = (float*)alloc(S * S * 4);
  unsigned short* P_bf = (unsigned short*)alloc(S * S * 2);
  float* attn_f = (float*)alloc(S * (NH_N * VD_D) * 4);
  unsigned short* attn_bf = (unsigned short*)alloc(S * (NH_N * VD_D) * 2);

  auto eb = [](long n) { return dim3((unsigned)((n + 255) / 256)); };
  auto gg = [](int N, int M) { return dim3((unsigned)((N + 127) / 128), (unsigned)((M + 127) / 128)); };

  // ---- bf16 casts / weight transposes ----
  cast_bf16_kernel<<<eb((long)S * H_DIM), 256, 0, stream>>>(hidden, hidden_bf, (int)(S * H_DIM));
  transpose_cast_kernel<<<eb((long)H_DIM * QL_DIM), 256, 0, stream>>>(Wq_a, WqaT, H_DIM, QL_DIM);
  transpose_cast_kernel<<<eb((long)QL_DIM * QB_DIM), 256, 0, stream>>>(Wq_b, WqbT, QL_DIM, QB_DIM);
  transpose_cast_kernel<<<eb((long)H_DIM * 576), 256, 0, stream>>>(Wkv_a, WkvaT, H_DIM, 576);
  transpose_cast_kernel<<<eb((long)KVL_DIM * KVB_DIM), 256, 0, stream>>>(Wkv_b, WkvbT, KVL_DIM, KVB_DIM);
  transpose_cast_kernel<<<eb((long)(NH_N * VD_D) * H_DIM), 256, 0, stream>>>(Wo, WoT, NH_N * VD_D, H_DIM);
  transpose_cast_kernel<<<eb((long)QL_DIM * (IH_N * ID_D)), 256, 0, stream>>>(Wq_idx, WqiT, QL_DIM, IH_N * ID_D);
  transpose_cast_kernel<<<eb((long)H_DIM * ID_D), 256, 0, stream>>>(Wk_idx, WkiT, H_DIM, ID_D);
  transpose_cast_kernel<<<eb((long)H_DIM * IH_N), 256, 0, stream>>>(Ww_idx, WwiT, H_DIM, IH_N);

  // ---- projection GEMMs (all WMMA bf16) ----
  gemm_nt_bf16<<<gg(QL_DIM, S_LEN), 256, 0, stream>>>(hidden_bf, WqaT, qa_f,
      S_LEN, QL_DIM, H_DIM, H_DIM, H_DIM, QL_DIM, 1.0f);
  rmsnorm_bf16_kernel<<<dim3(S_LEN), 256, 0, stream>>>(qa_f, q_a_g, qa_bf, QL_DIM, QL_DIM, QL_DIM);
  gemm_nt_bf16<<<gg(QB_DIM, S_LEN), 256, 0, stream>>>(qa_bf, WqbT, q_f,
      S_LEN, QB_DIM, QL_DIM, QL_DIM, QL_DIM, QB_DIM, 1.0f);
  gemm_nt_bf16<<<gg(576, S_LEN), 256, 0, stream>>>(hidden_bf, WkvaT, kv_f,
      S_LEN, 576, H_DIM, H_DIM, H_DIM, 576, 1.0f);
  rmsnorm_bf16_kernel<<<dim3(S_LEN), 256, 0, stream>>>(kv_f, kv_a_g, kvc_bf, KVL_DIM, 576, KVL_DIM);
  gemm_nt_bf16<<<gg(KVB_DIM, S_LEN), 256, 0, stream>>>(kvc_bf, WkvbT, kvb_f,
      S_LEN, KVB_DIM, KVL_DIM, KVL_DIM, KVL_DIM, KVB_DIM, 1.0f);
  gemm_nt_bf16<<<gg(IH_N * ID_D, S_LEN), 256, 0, stream>>>(qa_bf, WqiT, qi_f,
      S_LEN, IH_N * ID_D, QL_DIM, QL_DIM, QL_DIM, IH_N * ID_D, 1.0f);
  gemm_nt_bf16<<<gg(ID_D, S_LEN), 256, 0, stream>>>(hidden_bf, WkiT, ki_f,
      S_LEN, ID_D, H_DIM, H_DIM, H_DIM, ID_D, 1.0f);
  gemm_nt_bf16<<<gg(IH_N, S_LEN), 256, 0, stream>>>(hidden_bf, WwiT, w_f,
      S_LEN, IH_N, H_DIM, H_DIM, H_DIM, IH_N, 1.0f);

  // ---- build rope'd bf16 operands ----
  build_q_kernel<<<eb((long)NH_N * S * QKD), 256, 0, stream>>>(q_f, cosb, sinb, Qcat);
  build_k_kernel<<<eb((long)NH_N * S * QKD), 256, 0, stream>>>(kvb_f, kv_f, cosb, sinb, Kcat);
  build_vt_kernel<<<eb((long)NH_N * VD_D * S), 256, 0, stream>>>(kvb_f, Vt);
  build_qi_kernel<<<eb((long)IH_N * S * ID_D), 256, 0, stream>>>(qi_f, cosb, sinb, Qib);
  ln_rope_ki_kernel<<<dim3(S_LEN), 128, 0, stream>>>(ki_f, kn_g, kn_b, cosb, sinb, Kib);

  // ---- indexer + top-k selection ----
  idx_scores_kernel<<<dim3(S_LEN / 64, S_LEN / 128), 256, 0, stream>>>(Qib, Kib, w_f, idx_sc);
  topk_mask_kernel<<<dim3(S_LEN), 256, 0, stream>>>(idx_sc, mask, TOPK_N);

  // ---- per-head sparse attention ----
  const float smx_scale = 0.07216878364870323f;  // (NOPE+ROPE)^-0.5
  for (int h = 0; h < NH_N; ++h) {
    const unsigned short* Qh = Qcat + (size_t)h * S * QKD;
    const unsigned short* Kh = Kcat + (size_t)h * S * QKD;
    const unsigned short* Vh = Vt + (size_t)h * VD_D * S;
    gemm_nt_bf16<<<gg(S_LEN, S_LEN), 256, 0, stream>>>(Qh, Kh, sc_tmp,
        S_LEN, S_LEN, QKD, QKD, QKD, S_LEN, smx_scale);
    softmax_mask_kernel<<<dim3(S_LEN), 256, 0, stream>>>(sc_tmp, mask, P_bf);
    gemm_nt_bf16<<<gg(VD_D, S_LEN), 256, 0, stream>>>(P_bf, Vh,
        attn_f + (size_t)h * VD_D, S_LEN, VD_D, S_LEN, S_LEN, S_LEN,
        NH_N * VD_D, 1.0f);
  }

  // ---- output projection ----
  cast_bf16_kernel<<<eb((long)S * NH_N * VD_D), 256, 0, stream>>>(
      attn_f, attn_bf, (int)(S * NH_N * VD_D));
  gemm_nt_bf16<<<gg(H_DIM, S_LEN), 256, 0, stream>>>(attn_bf, WoT, out_final,
      S_LEN, H_DIM, NH_N * VD_D, NH_N * VD_D, NH_N * VD_D, H_DIM, 1.0f);
}
